// NodeModel_6536940224659
// MI455X (gfx1250) — matrix operand order repaired and checked by reading, not compiled, not run on hardware
//
#include <hip/hip_runtime.h>

typedef __attribute__((ext_vector_type(16))) _Float16 v16h;
typedef __attribute__((ext_vector_type(8)))  float    v8f;

#define WAVES 8
#define TPB   256

__device__ __forceinline__ v8f wmma16(v16h a, v16h b, v8f c) {
  // D = A(16x32,f16) * B(32x16,f16) + C(16x16,f32)
  return __builtin_amdgcn_wmma_f32_16x16x32_f16(false, a, false, b, (short)0, c,
                                                false, false);
}

// Load one B fragment (32 bytes per lane) from LDS fragment storage.
__device__ __forceinline__ v16h ld_frag(const unsigned int* fragbase, int lane) {
  union { uint4 q[2]; v16h v; } u;
  const uint4* p = (const uint4*)(fragbase + lane * 8);
  u.q[0] = p[0];
  u.q[1] = p[1];
  return u.v;
}

// 8 consecutive f32 from global -> 8 f16
__device__ __forceinline__ void cvt8(const float* p, _Float16* d) {
  float4 a = ((const float4*)p)[0];
  float4 b = ((const float4*)p)[1];
  d[0] = (_Float16)a.x; d[1] = (_Float16)a.y; d[2] = (_Float16)a.z; d[3] = (_Float16)a.w;
  d[4] = (_Float16)b.x; d[5] = (_Float16)b.y; d[6] = (_Float16)b.z; d[7] = (_Float16)b.w;
}

// A fragment from an LDS f16 row (two 16B chunks per lane)
__device__ __forceinline__ v16h ldA_lds(const _Float16* rowp, int kbase, int koff) {
  union { uint4 q[2]; v16h v; } u;
  u.q[0] = *(const uint4*)(rowp + kbase + koff);
  u.q[1] = *(const uint4*)(rowp + kbase + 16 + koff);
  return u.v;
}

// ---------------------------------------------------------------------------
// Kernel 1: per-edge message MLP + scatter-add into agg (= d_out)
//   h = x[row];  msg = relu(h@W1+b1)@W2+b2;  agg[col] += msg
// One wave handles a tile of 16 edges.
// ---------------------------------------------------------------------------
__global__ __launch_bounds__(TPB) void edge_msg_kernel(
    const float* __restrict__ x,
    const int*   __restrict__ eidx,   // [2, E]
    const float* __restrict__ w1, const float* __restrict__ b1,
    const float* __restrict__ w2, const float* __restrict__ b2,
    float* __restrict__ agg, int E, int niter)
{
  // 16 fragments (w1: kt0..1 x nt0..3 -> f=0..7, w2 -> f=8..15), 1KB each
  __shared__ alignas(16) unsigned int wfrag[16 * 32 * 8];
  __shared__ alignas(16) _Float16 mbuf[WAVES][16 * 64];
  __shared__ float b1s[64], b2s[64];
  __shared__ int colbuf[WAVES][16];

  const int tid = threadIdx.x;

  // Stage f16 weight fragments in B-fragment lane layout.
  for (int i = tid; i < 16 * 32 * 8; i += TPB) {
    int f = i >> 8, rem = i & 255, L = rem >> 3, v = rem & 7;
    const float* W = (f < 8) ? w1 : w2;
    int ff = f & 7, kt = ff >> 2, nt = ff & 3;
    int k = kt * 32 + ((L & 16) ? 8 : 0) + 2 * v + ((v & 4) ? 8 : 0);
    int n = nt * 16 + (L & 15);
    union { _Float16 h[2]; unsigned int u; } p;
    p.h[0] = (_Float16)W[k * 64 + n];
    p.h[1] = (_Float16)W[(k + 1) * 64 + n];
    wfrag[i] = p.u;
  }
  if (tid < 64) { b1s[tid] = b1[tid]; b2s[tid] = b2[tid]; }
  __syncthreads();

  const int wave = tid >> 5, lane = tid & 31;
  const int m = lane & 15, hi = lane >> 4, koff = hi * 8;
  const int tiles = (E + 15) >> 4;
  const int twaves = gridDim.x * WAVES;
  const int* rowI = eidx;
  const int* colI = eidx + E;
  int t = blockIdx.x * WAVES + wave;

  for (int it = 0; it < niter; ++it, t += twaves) {
    bool active = (t < tiles);
    if (active) {
      int base = t * 16;
      int e = base + m; if (e >= E) e = E - 1;
      int row = rowI[e];
      if (lane < 16) colbuf[wave][m] = colI[e];
      const float* src = x + (size_t)row * 64;

      v16h a0, a1;
      { union { _Float16 e16[16]; v16h v; } u;
        cvt8(src + koff,      u.e16);
        cvt8(src + 16 + koff, u.e16 + 8);
        a0 = u.v; }
      { union { _Float16 e16[16]; v16h v; } u;
        cvt8(src + 32 + koff, u.e16);
        cvt8(src + 48 + koff, u.e16 + 8);
        a1 = u.v; }

      _Float16* mb = mbuf[wave];
#pragma unroll
      for (int nt = 0; nt < 4; ++nt) {
        float bv = b1s[nt * 16 + m];
        v8f c = {bv, bv, bv, bv, bv, bv, bv, bv};
        c = wmma16(a0, ld_frag(&wfrag[(0 * 4 + nt) * 256], lane), c);
        c = wmma16(a1, ld_frag(&wfrag[(1 * 4 + nt) * 256], lane), c);
#pragma unroll
        for (int r = 0; r < 8; ++r) {
          float v = c[r] > 0.f ? c[r] : 0.f;
          mb[(r + hi * 8) * 64 + nt * 16 + m] = (_Float16)v;   // C-layout -> row-major LDS
        }
      }
    }
    __syncthreads();
    if (active) {
      int base = t * 16;
      const _Float16* mb = mbuf[wave] + m * 64;       // this lane's A row
      v16h a0 = ldA_lds(mb, 0,  koff);
      v16h a1 = ldA_lds(mb, 32, koff);
#pragma unroll
      for (int nt = 0; nt < 4; ++nt) {
        float bv = b2s[nt * 16 + m];
        v8f c = {bv, bv, bv, bv, bv, bv, bv, bv};
        c = wmma16(a0, ld_frag(&wfrag[(8 + nt) * 256],  lane), c);
        c = wmma16(a1, ld_frag(&wfrag[(12 + nt) * 256], lane), c);
#pragma unroll
        for (int r = 0; r < 8; ++r) {
          int M = r + hi * 8;
          if (base + M < E) {
            int col = colbuf[wave][M];
            atomicAdd(&agg[(size_t)col * 64 + nt * 16 + m], c[r]);
          }
        }
      }
    }
    __syncthreads();
  }
}

// ---------------------------------------------------------------------------
// Kernel 2: per-node output MLP:
//   out = relu(concat(x, agg) @ OW1 + ob1) @ OW2 + ob2
// agg lives in `out` (in-place safe: each tile owns disjoint node rows).
// ---------------------------------------------------------------------------
__global__ __launch_bounds__(TPB) void node_out_kernel(
    const float* __restrict__ x,
    const float* __restrict__ w1 /*128x64*/, const float* __restrict__ b1,
    const float* __restrict__ w2 /*64x64*/,  const float* __restrict__ b2,
    float* __restrict__ out, int N, int niter)
{
  // w1: kt0..3 x nt0..3 -> f=0..15 ; w2: f=16..23
  __shared__ alignas(16) unsigned int wfrag[24 * 32 * 8];
  __shared__ alignas(16) _Float16 mbuf[WAVES][16 * 64];
  __shared__ float b1s[64], b2s[64];

  const int tid = threadIdx.x;
  for (int i = tid; i < 24 * 32 * 8; i += TPB) {
    int f = i >> 8, rem = i & 255, L = rem >> 3, v = rem & 7;
    const float* W; int kt, nt;
    if (f < 16) { W = w1; kt = f >> 2; nt = f & 3; }
    else        { int f2 = f - 16; W = w2; kt = f2 >> 2; nt = f2 & 3; }
    int k = kt * 32 + ((L & 16) ? 8 : 0) + 2 * v + ((v & 4) ? 8 : 0);
    int n = nt * 16 + (L & 15);
    union { _Float16 h[2]; unsigned int u; } p;
    p.h[0] = (_Float16)W[k * 64 + n];
    p.h[1] = (_Float16)W[(k + 1) * 64 + n];
    wfrag[i] = p.u;
  }
  if (tid < 64) { b1s[tid] = b1[tid]; b2s[tid] = b2[tid]; }
  __syncthreads();

  const int wave = tid >> 5, lane = tid & 31;
  const int m = lane & 15, hi = lane >> 4, koff = hi * 8;
  const int tiles = (N + 15) >> 4;
  const int twaves = gridDim.x * WAVES;
  int t = blockIdx.x * WAVES + wave;

  for (int it = 0; it < niter; ++it, t += twaves) {
    bool active = (t < tiles);
    if (active) {
      int base = t * 16;
      int node = base + m; if (node >= N) node = N - 1;
      const float* sx = x   + (size_t)node * 64;
      const float* sg = out + (size_t)node * 64;   // agg (scatter result)

      v16h aX0, aX1, aG0, aG1;
      { union { _Float16 e16[16]; v16h v; } u;
        cvt8(sx + koff, u.e16); cvt8(sx + 16 + koff, u.e16 + 8); aX0 = u.v; }
      { union { _Float16 e16[16]; v16h v; } u;
        cvt8(sx + 32 + koff, u.e16); cvt8(sx + 48 + koff, u.e16 + 8); aX1 = u.v; }
      { union { _Float16 e16[16]; v16h v; } u;
        cvt8(sg + koff, u.e16); cvt8(sg + 16 + koff, u.e16 + 8); aG0 = u.v; }
      { union { _Float16 e16[16]; v16h v; } u;
        cvt8(sg + 32 + koff, u.e16); cvt8(sg + 48 + koff, u.e16 + 8); aG1 = u.v; }

      _Float16* mb = mbuf[wave];
#pragma unroll
      for (int nt = 0; nt < 4; ++nt) {
        float bv = b1s[nt * 16 + m];
        v8f c = {bv, bv, bv, bv, bv, bv, bv, bv};
        c = wmma16(aX0, ld_frag(&wfrag[(0 * 4 + nt) * 256], lane), c);  // K 0..31
        c = wmma16(aX1, ld_frag(&wfrag[(1 * 4 + nt) * 256], lane), c);  // K 32..63
        c = wmma16(aG0, ld_frag(&wfrag[(2 * 4 + nt) * 256], lane), c);  // K 64..95
        c = wmma16(aG1, ld_frag(&wfrag[(3 * 4 + nt) * 256], lane), c);  // K 96..127
#pragma unroll
        for (int r = 0; r < 8; ++r) {
          float v = c[r] > 0.f ? c[r] : 0.f;
          mb[(r + hi * 8) * 64 + nt * 16 + m] = (_Float16)v;
        }
      }
    }
    __syncthreads();
    if (active) {
      int base = t * 16;
      const _Float16* mb = mbuf[wave] + m * 64;
      v16h a0 = ldA_lds(mb, 0,  koff);
      v16h a1 = ldA_lds(mb, 32, koff);
#pragma unroll
      for (int nt = 0; nt < 4; ++nt) {
        float bv = b2s[nt * 16 + m];
        v8f c = {bv, bv, bv, bv, bv, bv, bv, bv};
        c = wmma16(a0, ld_frag(&wfrag[(16 + nt) * 256], lane), c);
        c = wmma16(a1, ld_frag(&wfrag[(20 + nt) * 256], lane), c);
#pragma unroll
        for (int r = 0; r < 8; ++r) {
          int node = base + r + hi * 8;
          if (node < N) out[(size_t)node * 64 + nt * 16 + m] = c[r];
        }
      }
    }
    __syncthreads();
  }
}

extern "C" void kernel_launch(void* const* d_in, const int* in_sizes, int n_in,
                              void* d_out, int out_size, void* d_ws, size_t ws_size,
                              hipStream_t stream) {
  const float* x      = (const float*)d_in[0];
  const int*   eidx   = (const int*)d_in[1];     // [2,E]
  const float* msg_w1 = (const float*)d_in[3];
  const float* msg_b1 = (const float*)d_in[4];
  const float* msg_w2 = (const float*)d_in[5];
  const float* msg_b2 = (const float*)d_in[6];
  const float* out_w1 = (const float*)d_in[7];
  const float* out_b1 = (const float*)d_in[8];
  const float* out_w2 = (const float*)d_in[9];
  const float* out_b2 = (const float*)d_in[10];
  float* out = (float*)d_out;

  const int N = in_sizes[0] / 64;
  const int E = in_sizes[1] / 2;

  // d_out doubles as the agg accumulator for the scatter-sum: zero it first.
  hipMemsetAsync(out, 0, (size_t)N * 64 * sizeof(float), stream);

  int tiles1 = (E + 15) / 16;
  int blocks1 = (tiles1 + WAVES - 1) / WAVES;
  if (blocks1 > 2048) blocks1 = 2048;
  int niter1 = (tiles1 + blocks1 * WAVES - 1) / (blocks1 * WAVES);
  edge_msg_kernel<<<blocks1, TPB, 0, stream>>>(x, eidx, msg_w1, msg_b1,
                                               msg_w2, msg_b2, out, E, niter1);

  int tiles2 = (N + 15) / 16;
  int blocks2 = (tiles2 + WAVES - 1) / WAVES;
  if (blocks2 > 1024) blocks2 = 1024;
  int niter2 = (tiles2 + blocks2 * WAVES - 1) / (blocks2 * WAVES);
  node_out_kernel<<<blocks2, TPB, 0, stream>>>(x, out_w1, out_b1,
                                               out_w2, out_b2, out, N, niter2);
}
